// GCN_LSTM_Model_11605001633904
// MI455X (gfx1250) — matrix operand order repaired and checked by reading, not compiled, not run on hardware
//
#include <hip/hip_runtime.h>
#include <hip/hip_bf16.h>
#include <math.h>

// ---------------------------------------------------------------------------
// Types for CDNA5 WMMA
// ---------------------------------------------------------------------------
typedef __attribute__((ext_vector_type(16))) _Float16 v16h;
typedef __attribute__((ext_vector_type(8)))  _Float16 v8h;
typedef __attribute__((ext_vector_type(8)))  float    v8f;

#define N_KP    17
#define SEQ_T   20
#define BATCH   1024
#define HIDDEN  256
#define GCN_OUT 128
#define LSTM_IN (GCN_OUT * N_KP)   // 2176

__device__ __forceinline__ float sigmoidf_(float x) {
    return 1.0f / (1.0f + __expf(-x));
}
__device__ __forceinline__ float tanhf_(float x) {
    // tanh(x) = 1 - 2/(exp(2x)+1)
    return 1.0f - 2.0f / (__expf(2.0f * x) + 1.0f);
}

#define SHUF16(lo, hi) __builtin_shufflevector(lo, hi, \
    0, 1, 2, 3, 4, 5, 6, 7, 8, 9, 10, 11, 12, 13, 14, 15)

// A fragment (16x32 f16): lanes 0-15 -> K[k..k+7],K[k+16..k+23];
//                         lanes 16-31 -> K[k+8..k+15],K[k+24..k+31]
__device__ __forceinline__ v16h load_afrag(const _Float16* __restrict__ arow,
                                           int k0, int hi) {
    const _Float16* p = arow + k0 + hi * 8;
    v8h lo = *(const v8h*)(p);
    v8h hh = *(const v8h*)(p + 16);
    return SHUF16(lo, hh);
}
// B fragment (32x16 f16) from [N,K] weights: lane holds column n,
// K[k0+hi*8..+7] then K[k0+16+hi*8..+7] (same order as A fragment).
__device__ __forceinline__ v16h load_bfrag(const _Float16* __restrict__ Wm,
                                           size_t n, int K, int k0, int hi) {
    const _Float16* p = Wm + n * K + k0 + hi * 8;
    v8h lo = *(const v8h*)(p);
    v8h hh = *(const v8h*)(p + 16);
    return SHUF16(lo, hh);
}

// ---------------------------------------------------------------------------
// Tiny utility kernels
// ---------------------------------------------------------------------------
__global__ void k_zero32(unsigned int* p, int n) {
    int i = blockIdx.x * blockDim.x + threadIdx.x;
    if (i < n) p[i] = 0u;
}

__global__ void k_cvt16(const float* __restrict__ s, _Float16* __restrict__ d, int n) {
    int i = blockIdx.x * blockDim.x + threadIdx.x;
    if (i < n) d[i] = (_Float16)s[i];
}

__global__ void k_addbias(const float* __restrict__ a, const float* __restrict__ b,
                          float* __restrict__ o, int n) {
    int i = blockIdx.x * blockDim.x + threadIdx.x;
    if (i < n) o[i] = a[i] + b[i];
}

// GCN normalized adjacency (17x17): tiny, single thread.
__global__ void k_adj(const int* __restrict__ e, float* __restrict__ A) {
    if (threadIdx.x != 0 || blockIdx.x != 0) return;
    int src[38 + N_KP], dst[38 + N_KP];
    float deg[N_KP], M[N_KP * N_KP];
    for (int i = 0; i < 38; ++i) { src[i] = e[i]; dst[i] = e[38 + i]; }
    for (int i = 0; i < N_KP; ++i) { src[38 + i] = i; dst[38 + i] = i; }
    for (int i = 0; i < N_KP; ++i) deg[i] = 0.0f;
    for (int i = 0; i < 38 + N_KP; ++i) deg[dst[i]] += 1.0f;
    for (int i = 0; i < N_KP; ++i) deg[i] = rsqrtf(deg[i]);
    for (int i = 0; i < N_KP * N_KP; ++i) M[i] = 0.0f;
    for (int i = 0; i < 38 + N_KP; ++i)
        M[dst[i] * N_KP + src[i]] += deg[src[i]] * deg[dst[i]];
    for (int i = 0; i < N_KP * N_KP; ++i) A[i] = M[i];
}

// ---------------------------------------------------------------------------
// GCN: per (b,t) sample, 17x3 -> 17x64 -> 17x128, output seq (f16, time-major)
// seq row index = t*BATCH + b, row length 2176.
// ---------------------------------------------------------------------------
__global__ __launch_bounds__(256)
void k_gcn(const float* __restrict__ x,
           const float* __restrict__ w1, const float* __restrict__ b1,
           const float* __restrict__ w2, const float* __restrict__ b2,
           const float* __restrict__ A,
           _Float16* __restrict__ seq) {
    const int bt = blockIdx.x;              // 0 .. B*T-1
    const int b  = bt & (BATCH - 1);
    const int t  = bt >> 10;
    const int tid = threadIdx.x, nt = blockDim.x;
    const float* xs = x + ((size_t)b * SEQ_T + t) * (N_KP * 3);

    __shared__ float As[N_KP * N_KP];
    __shared__ float p1[N_KP * 64];
    __shared__ float h1[N_KP * 64];
    __shared__ float p2[N_KP * GCN_OUT];

    for (int i = tid; i < N_KP * N_KP; i += nt) As[i] = A[i];
    __syncthreads();

    for (int i = tid; i < N_KP * 64; i += nt) {
        int m = i >> 6, j = i & 63;
        p1[i] = xs[m * 3 + 0] * w1[0 * 64 + j]
              + xs[m * 3 + 1] * w1[1 * 64 + j]
              + xs[m * 3 + 2] * w1[2 * 64 + j];
    }
    __syncthreads();
    for (int i = tid; i < N_KP * 64; i += nt) {
        int n = i >> 6, j = i & 63;
        float s = b1[j];
        #pragma unroll
        for (int m = 0; m < N_KP; ++m) s += As[n * N_KP + m] * p1[m * 64 + j];
        h1[i] = fmaxf(s, 0.0f);
    }
    __syncthreads();
    for (int i = tid; i < N_KP * GCN_OUT; i += nt) {
        int m = i >> 7, j = i & 127;
        float s = 0.0f;
        for (int k = 0; k < 64; ++k) s += h1[m * 64 + k] * w2[k * GCN_OUT + j];
        p2[i] = s;
    }
    __syncthreads();
    _Float16* out = seq + ((size_t)t * BATCH + b) * LSTM_IN;
    for (int i = tid; i < N_KP * GCN_OUT; i += nt) {
        int n = i >> 7, j = i & 127;
        float s = b2[j];
        #pragma unroll
        for (int m = 0; m < N_KP; ++m) s += As[n * N_KP + m] * p2[m * GCN_OUT + j];
        out[i] = (_Float16)fmaxf(s, 0.0f);
    }
}

// ---------------------------------------------------------------------------
// Big WMMA GEMM: C[M,N] = A[M,K] * W[N,K]^T + bias[N]  -> f16 out.
// One wave per 16x64 tile; 8 waves/block -> 128x64 block tile.
// Requires K % 64 == 0 (true for 2176 and 512 here).
// Ping-pong software pipeline: two fragment register sets alternate across
// even/odd K-chunks, so each set is written only by loads (no register
// copies), and next-chunk loads are always in flight behind the WMMAs.
// ---------------------------------------------------------------------------
__global__ __launch_bounds__(256)
void k_gemm_big(const _Float16* __restrict__ Am,   // [M,K]
                const _Float16* __restrict__ Wm,   // [N,K]
                const float*    __restrict__ bias, // [N]
                _Float16*       __restrict__ Of16, // [M,N]
                int M, int N, int K) {
    const int lane = threadIdx.x & 31;
    const int wave = threadIdx.x >> 5;
    const int m0 = blockIdx.y * 128 + wave * 16;
    const int n0 = blockIdx.x * 64;
    const int lm = lane & 15;
    const int hi = lane >> 4;

    v8f acc[4];
    #pragma unroll
    for (int s = 0; s < 4; ++s) {
        const float bv = bias[n0 + s * 16 + lm];
        #pragma unroll
        for (int r = 0; r < 8; ++r) acc[s][r] = bv;
    }

    const _Float16* arow = Am + (size_t)(m0 + lm) * K;

    // prologue: set0 <- chunk k=0
    v16h a0 = load_afrag(arow, 0, hi);
    v16h b0[4];
    #pragma unroll
    for (int s = 0; s < 4; ++s)
        b0[s] = load_bfrag(Wm, (size_t)(n0 + s * 16 + lm), K, 0, hi);

    v16h a1;
    v16h b1[4];
    int k0 = 0;
    for (; k0 + 64 < K; k0 += 64) {
        // set1 <- chunk k0+32 (issued before consuming set0)
        a1 = load_afrag(arow, k0 + 32, hi);
        #pragma unroll
        for (int s = 0; s < 4; ++s)
            b1[s] = load_bfrag(Wm, (size_t)(n0 + s * 16 + lm), K, k0 + 32, hi);
        #pragma unroll
        for (int s = 0; s < 4; ++s)
            acc[s] = __builtin_amdgcn_wmma_f32_16x16x32_f16(
                false, a0, false, b0[s], (short)0, acc[s], false, false);

        // set0 <- chunk k0+64 (issued before consuming set1)
        a0 = load_afrag(arow, k0 + 64, hi);
        #pragma unroll
        for (int s = 0; s < 4; ++s)
            b0[s] = load_bfrag(Wm, (size_t)(n0 + s * 16 + lm), K, k0 + 64, hi);
        #pragma unroll
        for (int s = 0; s < 4; ++s)
            acc[s] = __builtin_amdgcn_wmma_f32_16x16x32_f16(
                false, a1, false, b1[s], (short)0, acc[s], false, false);
    }
    // epilogue: chunks k0 (in set0) and k0+32
    a1 = load_afrag(arow, k0 + 32, hi);
    #pragma unroll
    for (int s = 0; s < 4; ++s)
        b1[s] = load_bfrag(Wm, (size_t)(n0 + s * 16 + lm), K, k0 + 32, hi);
    #pragma unroll
    for (int s = 0; s < 4; ++s)
        acc[s] = __builtin_amdgcn_wmma_f32_16x16x32_f16(
            false, a0, false, b0[s], (short)0, acc[s], false, false);
    #pragma unroll
    for (int s = 0; s < 4; ++s)
        acc[s] = __builtin_amdgcn_wmma_f32_16x16x32_f16(
            false, a1, false, b1[s], (short)0, acc[s], false, false);

    #pragma unroll
    for (int s = 0; s < 4; ++s)
        #pragma unroll
        for (int r = 0; r < 8; ++r)
            Of16[(size_t)(m0 + hi * 8 + r) * N + (n0 + s * 16 + lm)] =
                (_Float16)acc[s][r];
}

// ---------------------------------------------------------------------------
// Software grid barrier (per direction) for the persistent scan kernel.
// Two-variable (count/generation) barrier, agent-scope atomics, s_sleep spin.
// ---------------------------------------------------------------------------
__device__ __forceinline__ void gsync(unsigned* cnt, unsigned* gen, unsigned nb) {
    __syncthreads();
    if (threadIdx.x == 0) {
        unsigned g = __hip_atomic_load(gen, __ATOMIC_RELAXED, __HIP_MEMORY_SCOPE_AGENT);
        __threadfence();   // release all prior global writes
        unsigned prev = __hip_atomic_fetch_add(cnt, 1u, __ATOMIC_ACQ_REL,
                                               __HIP_MEMORY_SCOPE_AGENT);
        if (prev == nb - 1u) {
            __hip_atomic_store(cnt, 0u, __ATOMIC_RELAXED, __HIP_MEMORY_SCOPE_AGENT);
            __hip_atomic_fetch_add(gen, 1u, __ATOMIC_RELEASE,
                                   __HIP_MEMORY_SCOPE_AGENT);
        } else {
            while (__hip_atomic_load(gen, __ATOMIC_ACQUIRE,
                                     __HIP_MEMORY_SCOPE_AGENT) == g)
                __builtin_amdgcn_s_sleep(1);
        }
    }
    __syncthreads();
    __threadfence();       // acquire: discard stale WGP-cached h before reading
}

// ---------------------------------------------------------------------------
// Persistent fused bidirectional LSTM scan (one layer, both directions).
//   grid = (16 j-tiles, 4 m-blocks, 2 directions), block = 256 (8 waves).
//   Wave tile: rows [m0, m0+32), gate columns [j0, j0+16) for all 4 gates.
//   Per step: gates = gx[t] + h @ W_hh^T (WMMA, f32 acc), then the LSTM
//   nonlinearity entirely in registers. Cell state c lives in registers for
//   the whole scan (each wave exclusively owns its (m,j) tile). Hidden state
//   h is double-buffered in global f16 -> exactly one grid barrier per step.
// ---------------------------------------------------------------------------
__global__ __launch_bounds__(256)
void k_lstm_scan(_Float16* __restrict__ hF,        // [2][1024][256] ping-pong
                 _Float16* __restrict__ hR,        // [2][1024][256]
                 const _Float16* __restrict__ gxF, // [T][1024][1024]
                 const _Float16* __restrict__ gxR,
                 const _Float16* __restrict__ whF, // [1024][256]
                 const _Float16* __restrict__ whR,
                 _Float16* __restrict__ hseq,      // null or [T*B][512]
                 float* __restrict__ catout,       // null or [B][512]
                 unsigned* __restrict__ bar) {     // 4 u32: {cnt,gen} x 2 dirs
    const int dir = blockIdx.z;
    _Float16* hbuf = dir ? hR : hF;
    const _Float16* gx = dir ? gxR : gxF;
    const _Float16* wh = dir ? whR : whF;
    unsigned* cnt = bar + dir * 2;
    unsigned* gen = bar + dir * 2 + 1;
    const unsigned nb = gridDim.x * gridDim.y;   // blocks per direction (64)

    const int lane = threadIdx.x & 31;
    const int wave = threadIdx.x >> 5;
    const int lm = lane & 15;
    const int hi = lane >> 4;
    const int j0 = blockIdx.x * 16;              // hidden-column tile
    const int m0 = blockIdx.y * 256 + wave * 32; // rows [m0, m0+32)
    const size_t HB = (size_t)BATCH * HIDDEN;    // ping-pong stride

    v8f cacc[2] = {};                            // cell state, in registers

    // zero-init h (buffer 0); every (m,j) position covered exactly once
    #pragma unroll
    for (int u = 0; u < 2; ++u)
        #pragma unroll
        for (int r = 0; r < 8; ++r)
            hbuf[(size_t)(m0 + u * 16 + hi * 8 + r) * HIDDEN + j0 + lm] =
                (_Float16)0.0f;
    gsync(cnt, gen, nb);

    int p = 0;                                   // ping-pong phase
    for (int step = 0; step < SEQ_T; ++step) {
        const int t = dir ? (SEQ_T - 1 - step) : step;
        const _Float16* gxt = gx + (size_t)t * BATCH * 1024;
        const _Float16* hin = hbuf + (size_t)p * HB;
        _Float16* hout = hbuf + (size_t)(1 - p) * HB;

        // gate accumulators init from precomputed input projection
        v8f acc[2][4];
        #pragma unroll
        for (int u = 0; u < 2; ++u)
            #pragma unroll
            for (int g = 0; g < 4; ++g)
                #pragma unroll
                for (int r = 0; r < 8; ++r)
                    acc[u][g][r] = (float)gxt[(size_t)(m0 + u * 16 + hi * 8 + r)
                                              * 1024 + g * 256 + j0 + lm];

        // gates += h @ W_hh^T  (K = 256, fully unrolled)
        #pragma unroll
        for (int k0 = 0; k0 < HIDDEN; k0 += 32) {
            v16h a0 = load_afrag(hin + (size_t)(m0 + lm) * HIDDEN, k0, hi);
            v16h a1 = load_afrag(hin + (size_t)(m0 + 16 + lm) * HIDDEN, k0, hi);
            #pragma unroll
            for (int g = 0; g < 4; ++g) {
                v16h bf = load_bfrag(wh, (size_t)(g * 256 + j0 + lm),
                                     HIDDEN, k0, hi);
                acc[0][g] = __builtin_amdgcn_wmma_f32_16x16x32_f16(
                    false, a0, false, bf, (short)0, acc[0][g], false, false);
                acc[1][g] = __builtin_amdgcn_wmma_f32_16x16x32_f16(
                    false, a1, false, bf, (short)0, acc[1][g], false, false);
            }
        }

        // LSTM pointwise in registers (PyTorch gate order i,f,g,o)
        #pragma unroll
        for (int u = 0; u < 2; ++u) {
            #pragma unroll
            for (int r = 0; r < 8; ++r) {
                const float gi = sigmoidf_(acc[u][0][r]);
                const float gf = sigmoidf_(acc[u][1][r]);
                const float gg = tanhf_(acc[u][2][r]);
                const float go = sigmoidf_(acc[u][3][r]);
                const float cn = gf * cacc[u][r] + gi * gg;
                cacc[u][r] = cn;
                const float h = go * tanhf_(cn);
                const int row = m0 + u * 16 + hi * 8 + r;
                const int col = j0 + lm;
                hout[(size_t)row * HIDDEN + col] = (_Float16)h;
                if (hseq)
                    hseq[(size_t)(t * BATCH + row) * 512 + dir * 256 + col] =
                        (_Float16)h;
                if (catout && t == SEQ_T - 1)
                    catout[(size_t)row * 512 + dir * 256 + col] = h;
            }
        }

        gsync(cnt, gen, nb);   // make new h visible before next step reads it
        p ^= 1;
    }
}

// ---------------------------------------------------------------------------
// FC head (small, scalar)
// ---------------------------------------------------------------------------
__global__ __launch_bounds__(256)
void k_fc1(const float* __restrict__ cat, const float* __restrict__ w,
           const float* __restrict__ b, float* __restrict__ out) {
    const int idx = blockIdx.x * blockDim.x + threadIdx.x;  // 1024*64
    const int bi = idx >> 6, n = idx & 63;
    const float* cr = cat + (size_t)bi * 512;
    const float* wr = w + (size_t)n * 512;
    float s = b[n];
    for (int k = 0; k < 512; ++k) s += cr[k] * wr[k];
    out[idx] = fmaxf(s, 0.0f);
}

__global__ __launch_bounds__(256)
void k_fc2(const float* __restrict__ in, const float* __restrict__ w,
           const float* __restrict__ b, float* __restrict__ out) {
    const int idx = blockIdx.x * blockDim.x + threadIdx.x;  // 1024*12
    const int bi = idx / 12, n = idx % 12;
    float s = b[n];
    #pragma unroll
    for (int k = 0; k < 64; ++k) s += in[bi * 64 + k] * w[n * 64 + k];
    out[idx] = s;
}

// ---------------------------------------------------------------------------
// Host orchestration
// ---------------------------------------------------------------------------
extern "C" void kernel_launch(void* const* d_in, const int* in_sizes, int n_in,
                              void* d_out, int out_size, void* d_ws, size_t ws_size,
                              hipStream_t stream) {
    (void)in_sizes; (void)n_in; (void)out_size; (void)ws_size;

    const float* x       = (const float*)d_in[0];
    const float* gcn1_w  = (const float*)d_in[1];
    const float* gcn1_b  = (const float*)d_in[2];
    const float* gcn2_w  = (const float*)d_in[3];
    const float* gcn2_b  = (const float*)d_in[4];
    const float* w_ih[4] = { (const float*)d_in[5],  (const float*)d_in[9],
                             (const float*)d_in[13], (const float*)d_in[17] };
    const float* w_hh[4] = { (const float*)d_in[6],  (const float*)d_in[10],
                             (const float*)d_in[14], (const float*)d_in[18] };
    const float* b_ih[4] = { (const float*)d_in[7],  (const float*)d_in[11],
                             (const float*)d_in[15], (const float*)d_in[19] };
    const float* b_hh[4] = { (const float*)d_in[8],  (const float*)d_in[12],
                             (const float*)d_in[16], (const float*)d_in[20] };
    const float* fc1_w = (const float*)d_in[21];
    const float* fc1_b = (const float*)d_in[22];
    const float* fc2_w = (const float*)d_in[23];
    const float* fc2_b = (const float*)d_in[24];
    const int*   edge  = (const int*)d_in[25];

    // ---- workspace carve-up (256B aligned) ----
    char* ws = (char*)d_ws;
    size_t off = 0;
    auto alloc = [&](size_t bytes) -> char* {
        char* p = ws + off;
        off = (off + bytes + 255) & ~(size_t)255;
        return p;
    };
    const size_t MT = (size_t)BATCH * SEQ_T;  // 20480

    float*    adjA = (float*)alloc(N_KP * N_KP * sizeof(float));
    unsigned* bar  = (unsigned*)alloc(4 * sizeof(unsigned));
    float*    biasC[4];
    for (int i = 0; i < 4; ++i) biasC[i] = (float*)alloc(1024 * sizeof(float));
    _Float16* wih16[4];
    const int wih_n[4] = { 1024 * LSTM_IN, 1024 * LSTM_IN, 1024 * 512, 1024 * 512 };
    for (int i = 0; i < 4; ++i) wih16[i] = (_Float16*)alloc((size_t)wih_n[i] * 2);
    _Float16* whh16[4];
    for (int i = 0; i < 4; ++i) whh16[i] = (_Float16*)alloc((size_t)1024 * HIDDEN * 2);
    _Float16* seq16 = (_Float16*)alloc(MT * LSTM_IN * 2);
    _Float16* gx0f  = (_Float16*)alloc(MT * 1024 * 2);
    _Float16* gx0r  = (_Float16*)alloc(MT * 1024 * 2);
    _Float16* l0buf = (_Float16*)alloc(MT * 512 * 2);
    _Float16* gx1f  = (_Float16*)alloc(MT * 1024 * 2);
    _Float16* gx1r  = (_Float16*)alloc(MT * 1024 * 2);
    _Float16* hF    = (_Float16*)alloc((size_t)2 * BATCH * HIDDEN * 2); // ping-pong
    _Float16* hR    = (_Float16*)alloc((size_t)2 * BATCH * HIDDEN * 2);
    float*    cat   = (float*)alloc((size_t)BATCH * 512 * sizeof(float));
    float*    fc1o  = (float*)alloc((size_t)BATCH * 64 * sizeof(float));

    auto blks = [](int n) { return (n + 255) / 256; };

    // ---- prep: barrier counters, adjacency, weight conversion, biases ----
    k_zero32<<<1, 32, 0, stream>>>(bar, 4);
    k_adj<<<1, 32, 0, stream>>>(edge, adjA);
    for (int i = 0; i < 4; ++i) {
        k_cvt16<<<blks(wih_n[i]), 256, 0, stream>>>(w_ih[i], wih16[i], wih_n[i]);
        k_cvt16<<<blks(1024 * HIDDEN), 256, 0, stream>>>(w_hh[i], whh16[i],
                                                         1024 * HIDDEN);
        k_addbias<<<4, 256, 0, stream>>>(b_ih[i], b_hh[i], biasC[i], 1024);
    }

    // ---- GCN -> seq16 (time-major [t][b][2176]) ----
    k_gcn<<<(int)MT, 256, 0, stream>>>(x, gcn1_w, gcn1_b, gcn2_w, gcn2_b,
                                       adjA, seq16);

    // ---- layer 0 input projections (WMMA): gx = seq @ W_ih^T + bias ----
    auto gemm_big = [&](const _Float16* A, const _Float16* W, const float* bias,
                        _Float16* out, int M, int N, int K) {
        dim3 g(N / 64, M / 128);
        k_gemm_big<<<g, 256, 0, stream>>>(A, W, bias, out, M, N, K);
    };
    gemm_big(seq16, wih16[0], biasC[0], gx0f, (int)MT, 1024, LSTM_IN);
    gemm_big(seq16, wih16[1], biasC[1], gx0r, (int)MT, 1024, LSTM_IN);

    // ---- layer 0: persistent fused bidirectional scan ----
    dim3 sgrid(HIDDEN / 16, BATCH / 256, 2);   // (16, 4, 2)
    k_lstm_scan<<<sgrid, 256, 0, stream>>>(hF, hR, gx0f, gx0r,
                                           whh16[0], whh16[1],
                                           l0buf, nullptr, bar);

    // ---- layer 1 input projections ----
    gemm_big(l0buf, wih16[2], biasC[2], gx1f, (int)MT, 1024, 512);
    gemm_big(l0buf, wih16[3], biasC[3], gx1r, (int)MT, 1024, 512);

    // ---- layer 1: persistent fused bidirectional scan (writes cat @ t=19) ----
    k_lstm_scan<<<sgrid, 256, 0, stream>>>(hF, hR, gx1f, gx1r,
                                           whh16[2], whh16[3],
                                           nullptr, cat, bar);

    // ---- FC head ----
    k_fc1<<<blks(BATCH * 64), 256, 0, stream>>>(cat, fc1_w, fc1_b, fc1o);
    k_fc2<<<blks(BATCH * 12), 256, 0, stream>>>(fc1o, fc2_w, fc2_b,
                                                (float*)d_out);
}
